// LongformerNetModel_5136780886129
// MI455X (gfx1250) — matrix operand order repaired and checked by reading, not compiled, not run on hardware
//
#include <hip/hip_runtime.h>

typedef _Float16 v16h __attribute__((ext_vector_type(16)));
typedef _Float16 v8h  __attribute__((ext_vector_type(8)));
typedef float    v8f  __attribute__((ext_vector_type(8)));
typedef unsigned int u32x4 __attribute__((ext_vector_type(4)));
typedef int          i32x8 __attribute__((ext_vector_type(8)));
typedef int          i32x4 __attribute__((ext_vector_type(4)));

#define H_DIM  768
#define NHEAD  12
#define HD     64
#define LAYERS 12
#define FF_DIM 3072
#define WIN    256
#define MCH    128
#define SEQ    2048
#define BATCH  4
#define MTOK   (BATCH*SEQ)

#if __has_builtin(__builtin_amdgcn_tensor_load_to_lds)
#define USE_TDM 1
#else
#define USE_TDM 0
#endif

__device__ __forceinline__ v16h cat8(v8h lo, v8h hi) {
  return __builtin_shufflevector(lo, hi, 0,1,2,3,4,5,6,7,8,9,10,11,12,13,14,15);
}

__device__ __forceinline__ float gelu_exact(float x) {
  return 0.5f * x * (1.0f + erff(x * 0.70710678118654752f));
}

// ---------------------------------------------------------------------------
// Weight convert + transpose: W[K,N] f32  ->  Wt[N,K] f16 (k-contiguous rows,
// TDM friendly, and matches the WMMA B-fragment K-contiguous layout).
// ---------------------------------------------------------------------------
__global__ __launch_bounds__(256) void k_wcvt(
    const float* __restrict__ W, _Float16* __restrict__ Wt, int K, int N)
{
  __shared__ float t[32][33];
  const int n0 = blockIdx.x * 32, k0 = blockIdx.y * 32;
  const int tid = threadIdx.x;
  const int c = tid & 31, r8 = tid >> 5;
#pragma unroll
  for (int j = 0; j < 4; ++j) {
    const int r = r8 + j * 8;
    t[r][c] = W[(size_t)(k0 + r) * N + n0 + c];
  }
  __syncthreads();
#pragma unroll
  for (int j = 0; j < 4; ++j) {
    const int r = r8 + j * 8;                       // output row (n)
    Wt[(size_t)(n0 + r) * K + k0 + c] = (_Float16)t[c][r];
  }
}

// ---------------------------------------------------------------------------
// GEMM: out[M,N] = act(A[M,K] @ W[K,N] + bias (+ residual)), W given as
// Wt[N,K] f16. 128x64 tile, 8 waves, each wave owns a 2x2 grid of 16x16 WMMA
// tiles (4 WMMAs / K-step). A: f32->f16 converted into LDS. B: TDM
// tensor_load_to_lds with LDS row padding 64B->80B (fallback: v8h copy).
// ACT/HAS_RES are compile-time: branch-free epilogue.
// ---------------------------------------------------------------------------
template <int ACT, bool HAS_RES>
__global__ __launch_bounds__(256) void k_gemm(
    const float* __restrict__ A, const _Float16* __restrict__ Wt,
    const float* __restrict__ bias, const float* __restrict__ residual,
    float* __restrict__ out, int M, int N, int K)
{
  __shared__ _Float16 As[128 * 32];  // [m][k] stride 32 (64B rows)
  __shared__ _Float16 Bs[64 * 40];   // [n][k] stride 40 (80B rows, 16B-aligned)

  const int tid  = threadIdx.x;
  const int wave = tid >> 5, lane = tid & 31;
  const int lrow = lane & 15, hi = lane >> 4;
  const int m0 = blockIdx.y * 128, n0 = blockIdx.x * 64;
  const int mg = wave & 3, ng = wave >> 2;          // wave's 2x2 tile group
  const int khalfA = hi ? 16 : 0;
  const int kstB   = hi ? 16 : 0;

  int ncol[2];
  v8f acc[2][2];
#pragma unroll
  for (int in = 0; in < 2; ++in) {
    ncol[in] = n0 + (ng * 2 + in) * 16 + lrow;
    const float bz = bias[ncol[in]];
#pragma unroll
    for (int im = 0; im < 2; ++im)
#pragma unroll
      for (int r = 0; r < 8; ++r) acc[im][in][r] = bz;
  }

  for (int kt = 0; kt < K; kt += 32) {
#if USE_TDM
    if (wave == 0) {
      // ---- Tensor Data Mover: 32(k) x 64(n) f16 tile of Wt -> Bs ----
      const unsigned lds = (unsigned)(uintptr_t)(&Bs[0]);
      const unsigned long long ga =
          (unsigned long long)(uintptr_t)(Wt + (size_t)n0 * K + kt);
      // D# group0: count=1 | lds_addr[63:32] | global_addr[120:64] | type=2
      const unsigned long long q0 = 1ull | ((unsigned long long)lds << 32);
      const unsigned long long q1 = (ga & ((1ull << 57) - 1)) | (2ull << 62);
      const u32x4 g0 = { (unsigned)q0, (unsigned)(q0 >> 32),
                         (unsigned)q1, (unsigned)(q1 >> 32) };
      // D# group1: data_size=2B, pad_enable, pad every 64B by 16B,
      // tensor_dim0=K, tensor_dim1=N, tile_dim0=32, tile_dim1=64, d0_stride=K
      const unsigned long long kk = (unsigned long long)(unsigned)K;
      const unsigned long long nn = (unsigned long long)(unsigned)N;
      const unsigned long long p0 = (1ull << 16) | (1ull << 20) |
                                    (3ull << 22) | (3ull << 25) |
                                    ((kk & 0xffffull) << 48);
      const unsigned long long p1 = (kk >> 16) | (nn << 16) | (32ull << 48);
      const unsigned long long p2 = 64ull | (kk << 32);
      const unsigned long long p3 = 0;
      const i32x8 g1 = { (int)(unsigned)p0, (int)(p0 >> 32),
                         (int)(unsigned)p1, (int)(p1 >> 32),
                         (int)(unsigned)p2, (int)(p2 >> 32),
                         (int)(unsigned)p3, (int)(p3 >> 32) };
      const i32x4 z4 = { 0, 0, 0, 0 };
      const i32x8 z8 = { 0, 0, 0, 0, 0, 0, 0, 0 };
      __builtin_amdgcn_tensor_load_to_lds(g0, g1, z4, z4, z8, 0);
      __builtin_amdgcn_s_wait_tensorcnt((short)0);
    }
#else
    {
      const int n = tid >> 2, k8 = (tid & 3) << 3;
      const v8h wv = *(const v8h*)(Wt + (size_t)(n0 + n) * K + kt + k8);
      *(v8h*)(&Bs[n * 40 + k8]) = wv;
    }
#endif
    // ---- A tile: 128x32 f32 -> f16 in LDS ----
#pragma unroll
    for (int j = 0; j < 4; ++j) {
      const int id = tid + j * 256;                 // 1024 float4s
      const int r = id >> 3, c4 = (id & 7) << 2;
      const float4 f = *(const float4*)(A + (size_t)(m0 + r) * K + kt + c4);
      _Float16* d = &As[r * 32 + c4];
      d[0] = (_Float16)f.x; d[1] = (_Float16)f.y;
      d[2] = (_Float16)f.z; d[3] = (_Float16)f.w;
    }
    __syncthreads();

    v16h af[2], bf[2];
#pragma unroll
    for (int im = 0; im < 2; ++im) {
      const _Float16* pa = &As[((mg * 2 + im) * 16 + lrow) * 32 + khalfA];
      af[im] = cat8(*(const v8h*)pa, *(const v8h*)(pa + 8));
    }
#pragma unroll
    for (int in = 0; in < 2; ++in) {
      const _Float16* pb = &Bs[((ng * 2 + in) * 16 + lrow) * 40 + kstB];
      bf[in] = cat8(*(const v8h*)pb, *(const v8h*)(pb + 8));
    }
#pragma unroll
    for (int im = 0; im < 2; ++im)
#pragma unroll
      for (int in = 0; in < 2; ++in)
        acc[im][in] = __builtin_amdgcn_wmma_f32_16x16x32_f16(
            false, af[im], false, bf[in], (short)0, acc[im][in], false, false);
    __syncthreads();
  }

  // ---- fused epilogue: residual + activation + store (branch-free) ----
#pragma unroll
  for (int im = 0; im < 2; ++im) {
#pragma unroll
    for (int r = 0; r < 8; ++r) {
      const int m = m0 + (mg * 2 + im) * 16 + r + hi * 8;
#pragma unroll
      for (int in = 0; in < 2; ++in) {
        float vv = acc[im][in][r];
        if constexpr (HAS_RES) vv += residual[(size_t)m * N + ncol[in]];
        if constexpr (ACT == 1) vv = tanhf(vv);
        if constexpr (ACT == 2) vv = gelu_exact(vv);
        out[(size_t)m * N + ncol[in]] = vv;
      }
    }
  }
}

static inline void launch_gemm(const float* A, const _Float16* Wt,
                               const float* bias, const float* residual,
                               float* out, int M, int N, int K, int act,
                               hipStream_t stream)
{
  const dim3 g(N / 64, M / 128);
  if (residual) {
    k_gemm<0, true><<<g, 256, 0, stream>>>(A, Wt, bias, residual, out, M, N, K);
  } else if (act == 1) {
    k_gemm<1, false><<<g, 256, 0, stream>>>(A, Wt, bias, nullptr, out, M, N, K);
  } else if (act == 2) {
    k_gemm<2, false><<<g, 256, 0, stream>>>(A, Wt, bias, nullptr, out, M, N, K);
  } else {
    k_gemm<0, false><<<g, 256, 0, stream>>>(A, Wt, bias, nullptr, out, M, N, K);
  }
}

// ---------------------------------------------------------------------------
// Sliding-window attention, flash style.
// ---------------------------------------------------------------------------
__global__ __launch_bounds__(256) void k_attn(
    const float* __restrict__ q, const float* __restrict__ k,
    const float* __restrict__ v, float* __restrict__ o)
{
  const int qb   = blockIdx.x;
  const int head = blockIdx.y;
  const int bz   = blockIdx.z;
  const int tid  = threadIdx.x;
  const int wave = tid >> 5, lane = tid & 31;
  const int lrow = lane & 15, hi = lane >> 4;
  const int khalfA = hi ? 16 : 0;
  const int kstB   = hi ? 16 : 0;

  __shared__ _Float16 Ks[32 * 64];     // [key][d]
  __shared__ _Float16 Vt[64 * 32];     // [d][key]
  __shared__ _Float16 Ps[8 * 16 * 32]; // per-wave P bounce buffer

  const size_t chanoff = (size_t)head * HD;
  const size_t tokbase = (size_t)bz * SEQ;
  const int wq0 = qb * 256 + wave * 32;

  v16h qf[2][2];
#pragma unroll
  for (int sub = 0; sub < 2; ++sub) {
    const int tok = qb * 256 + wave * 32 + sub * 16 + lrow;
    const float* qrow = q + (tokbase + tok) * H_DIM + chanoff;
#pragma unroll
    for (int ds = 0; ds < 2; ++ds) {
      const float* p0 = qrow + ds * 32 + khalfA;
      const float* p1 = p0 + 8;
      v8h lo, hh;
#pragma unroll
      for (int e = 0; e < 8; ++e) { lo[e] = (_Float16)p0[e]; hh[e] = (_Float16)p1[e]; }
      qf[sub][ds] = cat8(lo, hh);
    }
  }

  v8f accO[2][4];
  float mst[2][8], lst[2][8];
#pragma unroll
  for (int s = 0; s < 2; ++s) {
#pragma unroll
    for (int t = 0; t < 4; ++t)
#pragma unroll
      for (int r = 0; r < 8; ++r) accO[s][t][r] = 0.0f;
#pragma unroll
    for (int r = 0; r < 8; ++r) { mst[s][r] = -1e30f; lst[s][r] = 0.0f; }
  }

  for (int c = 0; c < 24; ++c) {
    const int kg0 = qb * 256 - WIN + c * 32;
    if (kg0 + 31 < 0 || kg0 >= SEQ) continue;   // uniform over block

#pragma unroll
    for (int i = 0; i < 2; ++i) {
      const int idx = tid * 2 + i;
      const int key = idx >> 4, d4 = (idx & 15) << 2;
      const int kg = kg0 + key;
      float4 fk = make_float4(0.f, 0.f, 0.f, 0.f);
      float4 fv = make_float4(0.f, 0.f, 0.f, 0.f);
      if (kg >= 0 && kg < SEQ) {
        const size_t off = (tokbase + kg) * H_DIM + chanoff + d4;
        fk = *(const float4*)(k + off);
        fv = *(const float4*)(v + off);
      }
      _Float16* dk = &Ks[key * 64 + d4];
      dk[0] = (_Float16)fk.x; dk[1] = (_Float16)fk.y;
      dk[2] = (_Float16)fk.z; dk[3] = (_Float16)fk.w;
      Vt[(d4 + 0) * 32 + key] = (_Float16)fv.x;
      Vt[(d4 + 1) * 32 + key] = (_Float16)fv.y;
      Vt[(d4 + 2) * 32 + key] = (_Float16)fv.z;
      Vt[(d4 + 3) * 32 + key] = (_Float16)fv.w;
    }
    __syncthreads();

    const bool wactive = !((kg0 + 31 < wq0 - WIN) || (kg0 > wq0 + 31 + WIN));
    if (wactive) {
#pragma unroll
      for (int sub = 0; sub < 2; ++sub) {
        v8f s0, s1;
#pragma unroll
        for (int r = 0; r < 8; ++r) { s0[r] = 0.0f; s1[r] = 0.0f; }
#pragma unroll
        for (int ds = 0; ds < 2; ++ds) {
          const _Float16* pk0 = &Ks[lrow * 64 + ds * 32 + kstB];
          const _Float16* pk1 = &Ks[(16 + lrow) * 64 + ds * 32 + kstB];
          const v16h b0 = cat8(*(const v8h*)pk0, *(const v8h*)(pk0 + 8));
          const v16h b1 = cat8(*(const v8h*)pk1, *(const v8h*)(pk1 + 8));
          s0 = __builtin_amdgcn_wmma_f32_16x16x32_f16(false, qf[sub][ds], false, b0, (short)0, s0, false, false);
          s1 = __builtin_amdgcn_wmma_f32_16x16x32_f16(false, qf[sub][ds], false, b1, (short)0, s1, false, false);
        }

        const int kga = kg0 + lrow;
        const int kgb = kg0 + 16 + lrow;
        float rowm[8];
#pragma unroll
        for (int r = 0; r < 8; ++r) {
          const int qg = wq0 + sub * 16 + r + hi * 8;
          float sa = s0[r] * 0.125f;
          float sb = s1[r] * 0.125f;
          const int da = kga - qg, db = kgb - qg;
          const bool va = (kga >= 0) && (kga < SEQ) && (da <= WIN) && (da >= -WIN);
          const bool vb = (kgb >= 0) && (kgb < SEQ) && (db <= WIN) && (db >= -WIN);
          sa = va ? sa : -1e30f;
          sb = vb ? sb : -1e30f;
          s0[r] = sa; s1[r] = sb;
          rowm[r] = fmaxf(sa, sb);
        }
#pragma unroll
        for (int d = 1; d < 16; d <<= 1)
#pragma unroll
          for (int r = 0; r < 8; ++r)
            rowm[r] = fmaxf(rowm[r], __shfl_xor(rowm[r], d, 32));

        float alpha[8], rsum[8];
#pragma unroll
        for (int r = 0; r < 8; ++r) {
          const float mn = fmaxf(fmaxf(mst[sub][r], rowm[r]), -1e20f);
          alpha[r] = __expf(mst[sub][r] - mn);
          mst[sub][r] = mn;
          const float p0 = __expf(s0[r] - mn);
          const float p1 = __expf(s1[r] - mn);
          s0[r] = p0; s1[r] = p1;
          rsum[r] = p0 + p1;
        }
#pragma unroll
        for (int d = 1; d < 16; d <<= 1)
#pragma unroll
          for (int r = 0; r < 8; ++r)
            rsum[r] += __shfl_xor(rsum[r], d, 32);
#pragma unroll
        for (int r = 0; r < 8; ++r)
          lst[sub][r] = lst[sub][r] * alpha[r] + rsum[r];
#pragma unroll
        for (int t = 0; t < 4; ++t)
#pragma unroll
          for (int r = 0; r < 8; ++r)
            accO[sub][t][r] *= alpha[r];

        _Float16* pw = &Ps[wave * 512];
#pragma unroll
        for (int r = 0; r < 8; ++r) {
          const int m = r + hi * 8;
          pw[m * 32 + lrow]      = (_Float16)s0[r];
          pw[m * 32 + 16 + lrow] = (_Float16)s1[r];
        }
        asm volatile("s_wait_dscnt 0" ::: "memory");  // in-wave LDS RAW
        const _Float16* pp = &Ps[wave * 512 + lrow * 32 + khalfA];
        const v16h pf = cat8(*(const v8h*)pp, *(const v8h*)(pp + 8));

#pragma unroll
        for (int t = 0; t < 4; ++t) {
          const _Float16* pv = &Vt[(t * 16 + lrow) * 32 + kstB];
          const v16h bv = cat8(*(const v8h*)pv, *(const v8h*)(pv + 8));
          accO[sub][t] = __builtin_amdgcn_wmma_f32_16x16x32_f16(false, pf, false, bv, (short)0, accO[sub][t], false, false);
        }
      }
    }
    __syncthreads();
  }

#pragma unroll
  for (int sub = 0; sub < 2; ++sub) {
#pragma unroll
    for (int r = 0; r < 8; ++r) {
      const float inv = 1.0f / fmaxf(lst[sub][r], 1e-20f);
      const int tok = qb * 256 + wave * 32 + sub * 16 + r + hi * 8;
      float* orow = o + (tokbase + tok) * H_DIM + chanoff;
#pragma unroll
      for (int t = 0; t < 4; ++t)
        orow[t * 16 + lrow] = accO[sub][t][r] * inv;
    }
  }
}

// ---------------------------------------------------------------------------
__global__ __launch_bounds__(256) void k_ln(
    const float* __restrict__ in, const float* __restrict__ pos,
    const float* __restrict__ emb, const float* __restrict__ g,
    const float* __restrict__ b, float* __restrict__ out)
{
  __shared__ float red[256];
  const int row = blockIdx.x;
  const int s = row & (SEQ - 1), bb = row >> 11;
  const int tid = threadIdx.x;
  const float* pin = in + (size_t)row * H_DIM;
  float x[3];
#pragma unroll
  for (int i = 0; i < 3; ++i) {
    const int c = tid + i * 256;
    float val = pin[c];
    if (pos) val += pos[(size_t)s * H_DIM + c] + emb[(size_t)bb * H_DIM + c];
    x[i] = val;
  }
  red[tid] = x[0] + x[1] + x[2];
  __syncthreads();
  for (int off = 128; off > 0; off >>= 1) {
    if (tid < off) red[tid] += red[tid + off];
    __syncthreads();
  }
  const float mean = red[0] * (1.0f / 768.0f);
  __syncthreads();
  float vs = 0.0f;
#pragma unroll
  for (int i = 0; i < 3; ++i) { const float d = x[i] - mean; vs += d * d; }
  red[tid] = vs;
  __syncthreads();
  for (int off = 128; off > 0; off >>= 1) {
    if (tid < off) red[tid] += red[tid + off];
    __syncthreads();
  }
  const float rstd = rsqrtf(red[0] * (1.0f / 768.0f) + 1e-5f);
  float* pout = out + (size_t)row * H_DIM;
#pragma unroll
  for (int i = 0; i < 3; ++i) {
    const int c = tid + i * 256;
    pout[c] = (x[i] - mean) * rstd * g[c] + b[c];
  }
}

// ---------------------------------------------------------------------------
__global__ __launch_bounds__(256) void k_temb(
    const int* __restrict__ ts, const float* __restrict__ w1,
    const float* __restrict__ b1, const float* __restrict__ w2,
    const float* __restrict__ b2, float* __restrict__ emb)
{
  __shared__ float e[MCH];
  __shared__ float hid[4 * MCH];
  const int bb = blockIdx.x, tid = threadIdx.x;
  const float t = (float)ts[bb];
  if (tid < MCH) {
    const int j = tid & 63;
    const float f = __expf(-logf(10000.0f) * (float)j * (1.0f / 64.0f));
    const float a = t * f;
    e[tid] = (tid < 64) ? cosf(a) : sinf(a);
  }
  __syncthreads();
  for (int ii = tid; ii < 4 * MCH; ii += 256) {
    float acc = b1[ii];
    for (int j = 0; j < MCH; ++j) acc += e[j] * w1[j * 4 * MCH + ii];
    hid[ii] = acc / (1.0f + __expf(-acc));
  }
  __syncthreads();
  for (int ii = tid; ii < H_DIM; ii += 256) {
    float acc = b2[ii];
    for (int j = 0; j < 4 * MCH; ++j) acc += hid[j] * w2[j * H_DIM + ii];
    emb[bb * H_DIM + ii] = acc;
  }
}

// ---------------------------------------------------------------------------
extern "C" void kernel_launch(void* const* d_in, const int* in_sizes, int n_in,
                              void* d_out, int out_size, void* d_ws, size_t ws_size,
                              hipStream_t stream)
{
  const float* x       = (const float*)d_in[0];
  const int*   ts      = (const int*)  d_in[1];
  const float* t_w1    = (const float*)d_in[2];
  const float* t_b1    = (const float*)d_in[3];
  const float* t_w2    = (const float*)d_in[4];
  const float* t_b2    = (const float*)d_in[5];
  const float* up_w1   = (const float*)d_in[6];
  const float* up_b1   = (const float*)d_in[7];
  const float* up_w2   = (const float*)d_in[8];
  const float* up_b2   = (const float*)d_in[9];
  const float* pos_emb = (const float*)d_in[10];
  const float* ln_g    = (const float*)d_in[11];
  const float* ln_b    = (const float*)d_in[12];
  const float* Wq      = (const float*)d_in[13];
  const float* bq      = (const float*)d_in[14];
  const float* Wk      = (const float*)d_in[15];
  const float* bk      = (const float*)d_in[16];
  const float* Wv      = (const float*)d_in[17];
  const float* bv      = (const float*)d_in[18];
  const float* Wo      = (const float*)d_in[19];
  const float* bo      = (const float*)d_in[20];
  const float* g1      = (const float*)d_in[21];
  const float* b1      = (const float*)d_in[22];
  const float* Wi      = (const float*)d_in[23];
  const float* bi      = (const float*)d_in[24];
  const float* Wo2     = (const float*)d_in[25];
  const float* bo2     = (const float*)d_in[26];
  const float* g2      = (const float*)d_in[27];
  const float* b2      = (const float*)d_in[28];
  const float* dn_w1   = (const float*)d_in[29];
  const float* dn_b1   = (const float*)d_in[30];
  const float* dn_w2   = (const float*)d_in[31];
  const float* dn_b2   = (const float*)d_in[32];
  float* out = (float*)d_out;

  // ws (floats): emb | h | tmp | big(M*FF) | wt (f16, per-layer weights)
  float* ws  = (float*)d_ws;
  float* emb = ws;
  float* h   = ws + 4096;
  float* tmp = h + (size_t)MTOK * H_DIM;
  float* big = tmp + (size_t)MTOK * H_DIM;
  _Float16* wt = (_Float16*)(big + (size_t)MTOK * FF_DIM);
  const size_t MH = (size_t)MTOK * H_DIM;
  const size_t HH = (size_t)H_DIM * H_DIM;
  _Float16* wtq  = wt;
  _Float16* wtk  = wt + HH;
  _Float16* wtv  = wt + 2 * HH;
  _Float16* wto  = wt + 3 * HH;
  _Float16* wti  = wt + 4 * HH;                       // [FF][H]
  _Float16* wto2 = wt + 4 * HH + (size_t)H_DIM * FF_DIM;  // [H][FF]

  k_temb<<<BATCH, 256, 0, stream>>>(ts, t_w1, t_b1, t_w2, t_b2, emb);

  // prelude: tanh(x @ up_w1 + b) @ up_w2 + b ; + pos + emb ; LN
  k_wcvt<<<dim3(H_DIM/32, MCH/32),   256, 0, stream>>>(up_w1, wtq, MCH, H_DIM);
  launch_gemm(x, wtq, up_b1, nullptr, tmp, MTOK, H_DIM, MCH, 1, stream);
  k_wcvt<<<dim3(H_DIM/32, H_DIM/32), 256, 0, stream>>>(up_w2, wtk, H_DIM, H_DIM);
  launch_gemm(tmp, wtk, up_b2, nullptr, big, MTOK, H_DIM, H_DIM, 0, stream);
  k_ln<<<MTOK, 256, 0, stream>>>(big, pos_emb, emb, ln_g, ln_b, h);

  const dim3 cHH(H_DIM/32, H_DIM/32);
  for (int l = 0; l < LAYERS; ++l) {
    const size_t woff  = (size_t)l * HH;
    const size_t boff  = (size_t)l * H_DIM;
    const size_t wioff = (size_t)l * H_DIM * FF_DIM;
    const size_t bioff = (size_t)l * FF_DIM;

    k_wcvt<<<cHH, 256, 0, stream>>>(Wq + woff, wtq, H_DIM, H_DIM);
    k_wcvt<<<cHH, 256, 0, stream>>>(Wk + woff, wtk, H_DIM, H_DIM);
    k_wcvt<<<cHH, 256, 0, stream>>>(Wv + woff, wtv, H_DIM, H_DIM);
    k_wcvt<<<cHH, 256, 0, stream>>>(Wo + woff, wto, H_DIM, H_DIM);
    k_wcvt<<<dim3(FF_DIM/32, H_DIM/32), 256, 0, stream>>>(Wi  + wioff, wti,  H_DIM, FF_DIM);
    k_wcvt<<<dim3(H_DIM/32, FF_DIM/32), 256, 0, stream>>>(Wo2 + wioff, wto2, FF_DIM, H_DIM);

    launch_gemm(h, wtq, bq + boff, nullptr, big,          MTOK, H_DIM, H_DIM, 0, stream);
    launch_gemm(h, wtk, bk + boff, nullptr, big + MH,     MTOK, H_DIM, H_DIM, 0, stream);
    launch_gemm(h, wtv, bv + boff, nullptr, big + 2 * MH, MTOK, H_DIM, H_DIM, 0, stream);

    k_attn<<<dim3(SEQ/256, NHEAD, BATCH), 256, 0, stream>>>(big, big + MH, big + 2 * MH, big + 3 * MH);

    launch_gemm(big + 3 * MH, wto, bo + boff, h, tmp, MTOK, H_DIM, H_DIM, 0, stream);
    k_ln<<<MTOK, 256, 0, stream>>>(tmp, nullptr, nullptr, g1 + boff, b1 + boff, h);

    launch_gemm(h,   wti,  bi  + bioff, nullptr, big, MTOK, FF_DIM, H_DIM,  2, stream);
    launch_gemm(big, wto2, bo2 + boff,  h,       tmp, MTOK, H_DIM,  FF_DIM, 0, stream);
    k_ln<<<MTOK, 256, 0, stream>>>(tmp, nullptr, nullptr, g2 + boff, b2 + boff, h);
  }

  k_wcvt<<<cHH, 256, 0, stream>>>(dn_w1, wtq, H_DIM, H_DIM);
  launch_gemm(h, wtq, dn_b1, nullptr, tmp, MTOK, H_DIM, H_DIM, 1, stream);
  k_wcvt<<<dim3(MCH/32, H_DIM/32), 256, 0, stream>>>(dn_w2, wtk, H_DIM, MCH);
  launch_gemm(tmp, wtk, dn_b2, nullptr, out, MTOK, MCH, H_DIM, 0, stream);

  (void)in_sizes; (void)n_in; (void)out_size; (void)ws_size;
}
